// ModernBertAttention_16200616640934
// MI455X (gfx1250) — matrix operand order repaired and checked by reading, not compiled, not run on hardware
//
#include <hip/hip_runtime.h>
#include <hip/hip_bf16.h>
#include <hip/hip_fp16.h>

// ---------------------------------------------------------------------------
// ModernBERT sliding-window attention for MI455X (gfx1250, wave32, WMMA).
//   B=1, S=4096, H=768, NH=12, HD=64, WINDOW=128 (|i-j| <= 64)
// Three stages, all built on v_wmma_f32_16x16x32_f16:
//   1) qkv = x @ Wqkv^T, fused RoPE, q pre-scaled by 1/sqrt(64), f16 ws
//   2) per-head sliding-window attention (Q@K^T, softmax, P@V), f16 ws
//   3) out = attn @ Wo^T, f32 output
// ---------------------------------------------------------------------------

typedef _Float16 half_t;
typedef __attribute__((ext_vector_type(16))) _Float16 v16h;
typedef __attribute__((ext_vector_type(8)))  _Float16 v8h;
typedef __attribute__((ext_vector_type(8)))  float    v8f;
typedef __attribute__((ext_vector_type(4)))  float    v4f;

constexpr int S  = 4096;
constexpr int H  = 768;
constexpr int NH = 12;
constexpr int HD = 64;
constexpr int QB    = 32;           // query block in attention kernel
constexpr int KSPAN = QB + 128;     // 160 keys cover the +-64 window

// ---- WMMA fragment helpers (16-bit A 16x32, B 32x16, f32 C 16x16) ---------

__device__ __forceinline__ v8f wmma_f16(v16h a, v16h b, v8f c) {
  // (neg_a, A, neg_b, B, c_mod, C, reuse_a, reuse_b)
  return __builtin_amdgcn_wmma_f32_16x16x32_f16(false, a, false, b,
                                                (short)0, c, false, false);
}

// A-matrix 16x32 (MxK), row-major source with stride (halfs).
// lane l (l0=l&15, hi=l>>4): elem j = K(kbase+hi*16+j), j=0..15.
__device__ __forceinline__ v16h load_a_frag(const half_t* base, int stride,
                                            int row0, int kbase) {
  const int lane = threadIdx.x & 31;
  const int l0 = lane & 15, hi = lane >> 4;
  const half_t* r = base + (row0 + l0) * stride + kbase + hi * 16;
  v8h lo = *(const v8h*)(r);
  v8h hv = *(const v8h*)(r + 8);
  v16h a;
#pragma unroll
  for (int j = 0; j < 8; ++j) { a[j] = lo[j]; a[8 + j] = hv[j]; }
  return a;
}

// B-matrix 32x16 (KxN) where source is stored "N rows x K cols" (i.e. B^T
// row-major, K contiguous): b[j] = BT[n0+l0][kbase + hi*16 + j].
__device__ __forceinline__ v16h load_bt_frag(const half_t* base, int stride,
                                             int n0, int kbase) {
  const int lane = threadIdx.x & 31;
  const int l0 = lane & 15, hi = lane >> 4;
  const half_t* r = base + (n0 + l0) * stride + kbase + hi * 16;
  v8h b0 = *(const v8h*)(r);
  v8h b1 = *(const v8h*)(r + 8);
  v16h b;
#pragma unroll
  for (int j = 0; j < 8; ++j) { b[j] = b0[j]; b[8 + j] = b1[j]; }
  return b;
}

// C/D f32 16x16: VGPR r holds M = hi*8 + r, N = l0.
__device__ __forceinline__ void store_c_lds(float* base, int stride,
                                            int m0, int n0, v8f c) {
  const int lane = threadIdx.x & 31;
  const int l0 = lane & 15, hi = lane >> 4;
#pragma unroll
  for (int r = 0; r < 8; ++r)
    base[(m0 + hi * 8 + r) * stride + n0 + l0] = c[r];
}

// ---------------------------------------------------------------------------
// Kernel 1: qkv = x @ Wqkv^T (4096x2304), fused RoPE, f16 per-head output.
// Grid (S/64, 2304/64), 256 threads. LDS: A 64x32 f16, B 64x32 f16, C 64x64 f32.
// ---------------------------------------------------------------------------
__global__ void __launch_bounds__(256) qkv_rope_kernel(
    const float* __restrict__ x, const float* __restrict__ Wqkv,
    const float* __restrict__ cosT, const float* __restrict__ sinT,
    half_t* __restrict__ qkvh) {
  extern __shared__ char smem[];
  half_t* At = (half_t*)smem;            // 64*32 f16
  half_t* Bt = At + 64 * 32;             // 64*32 f16
  float*  Ct = (float*)(Bt + 64 * 32);   // 64*64 f32

  const int m0 = blockIdx.x * 64;
  const int n0 = blockIdx.y * 64;
  const int t  = threadIdx.x;
  const int wid = t >> 5;
  const int wm = wid >> 1, wn = wid & 1;

  const int lrow = t >> 2;               // 0..63
  const int lcol = (t & 3) * 8;          // 0,8,16,24

  v8f acc0 = {}; v8f acc1 = {};
  for (int kc = 0; kc < H; kc += 32) {
    __syncthreads();
    {
      const float* sa = x + (size_t)(m0 + lrow) * H + kc + lcol;
      v4f a0 = *(const v4f*)sa;
      v4f a1 = *(const v4f*)(sa + 4);
      half_t* da = At + lrow * 32 + lcol;
#pragma unroll
      for (int j = 0; j < 4; ++j) { da[j] = (half_t)a0[j]; da[4 + j] = (half_t)a1[j]; }
      const float* sb = Wqkv + (size_t)(n0 + lrow) * H + kc + lcol;
      v4f b0 = *(const v4f*)sb;
      v4f b1 = *(const v4f*)(sb + 4);
      half_t* db = Bt + lrow * 32 + lcol;
#pragma unroll
      for (int j = 0; j < 4; ++j) { db[j] = (half_t)b0[j]; db[4 + j] = (half_t)b1[j]; }
      if (kc + 32 < H) {                 // emits global_prefetch_b8
        __builtin_prefetch(sa + 32, 0, 0);
        __builtin_prefetch(sb + 32, 0, 0);
      }
    }
    __syncthreads();
    v16h a  = load_a_frag(At, 32, wm * 16, 0);
    v16h b0 = load_bt_frag(Bt, 32, wn * 32, 0);
    v16h b1 = load_bt_frag(Bt, 32, wn * 32 + 16, 0);
    acc0 = wmma_f16(a, b0, acc0);
    acc1 = wmma_f16(a, b1, acc1);
  }
  __syncthreads();
  store_c_lds(Ct, 64, wm * 16, wn * 32, acc0);
  store_c_lds(Ct, 64, wm * 16, wn * 32 + 16, acc1);
  __syncthreads();

  // RoPE + writeout. One 64-wide N tile == exactly one (part, head) slot.
  const int part = n0 / H;               // 0=q, 1=k, 2=v
  const int head = (n0 % H) / HD;
  const int row = t >> 2;
  const int c0  = (t & 3) * 16;
  const int sg  = m0 + row;
  half_t* dst = qkvh + (((size_t)part * NH + head) * S + sg) * HD;
  if (part < 2) {
    const float* cr = cosT + (size_t)sg * HD;
    const float* sr = sinT + (size_t)sg * HD;
    const float scale = (part == 0) ? 0.125f : 1.0f;   // fold 1/sqrt(HD) into q
#pragma unroll
    for (int j = 0; j < 16; ++j) {
      const int d = c0 + j;
      const float v   = Ct[row * 64 + d];
      const float rot = (d < 32) ? -Ct[row * 64 + d + 32] : Ct[row * 64 + d - 32];
      dst[d] = (half_t)((v * cr[d] + rot * sr[d]) * scale);
    }
  } else {
#pragma unroll
    for (int j = 0; j < 16; ++j) {
      const int d = c0 + j;
      dst[d] = (half_t)Ct[row * 64 + d];
    }
  }
}

// ---------------------------------------------------------------------------
// Kernel 2: sliding-window attention. Grid (S/QB, NH), 256 threads.
// LDS: Q 32x64 f16 (4K) + K 160x64 f16 (20K) + V^T 64x160 f16 (20K)
//    + scores/probs 32x160 f16 (10K) = 54 KB.
// ---------------------------------------------------------------------------
__global__ void __launch_bounds__(256) attn_kernel(
    const half_t* __restrict__ qkvh, half_t* __restrict__ attn) {
  extern __shared__ char smem[];
  half_t* Qt = (half_t*)smem;              // [QB][HD]
  half_t* Kt = Qt + QB * HD;               // [KSPAN][HD]
  half_t* Vt = Kt + KSPAN * HD;            // [HD][KSPAN]  (transposed)
  half_t* Sc = Vt + HD * KSPAN;            // [QB][KSPAN]  scores -> probs

  const int q0  = blockIdx.x * QB;
  const int h   = blockIdx.y;
  const int kg0 = q0 - 64;
  const int t   = threadIdx.x;
  const int wid = t >> 5;

  const half_t* qh = qkvh + ((size_t)h * S) * HD;
  const half_t* kh = qkvh + ((size_t)(NH + h) * S) * HD;
  const half_t* vh = qkvh + ((size_t)(2 * NH + h) * S) * HD;

  // Stage Q: 32x64 halfs, 8 per thread.
  {
    const int row = t >> 3;
    const int c0 = (t & 7) * 8;
    *(v8h*)(Qt + row * HD + c0) = *(const v8h*)(qh + (size_t)(q0 + row) * HD + c0);
  }
  // Stage K (zero-pad out-of-range keys).
  for (int base = 0; base < KSPAN; base += 64) {
    const int row = base + (t >> 2);
    if (row < KSPAN) {
      const int c0 = (t & 3) * 16;
      const int kg = kg0 + row;
      if (kg >= 0 && kg < S) {
        *(v8h*)(Kt + row * HD + c0)     = *(const v8h*)(kh + (size_t)kg * HD + c0);
        *(v8h*)(Kt + row * HD + c0 + 8) = *(const v8h*)(kh + (size_t)kg * HD + c0 + 8);
      } else {
        v8h z = {};
        *(v8h*)(Kt + row * HD + c0)     = z;
        *(v8h*)(Kt + row * HD + c0 + 8) = z;
      }
    }
  }
  // Stage V transposed: Vt[d][key].
  for (int base = 0; base < KSPAN; base += 64) {
    const int key = base + (t >> 2);
    if (key < KSPAN) {
      const int c0 = (t & 3) * 16;
      const int kg = kg0 + key;
      if (kg >= 0 && kg < S) {
        v8h a = *(const v8h*)(vh + (size_t)kg * HD + c0);
        v8h b = *(const v8h*)(vh + (size_t)kg * HD + c0 + 8);
#pragma unroll
        for (int j = 0; j < 8; ++j) {
          Vt[(c0 + j) * KSPAN + key]     = a[j];
          Vt[(c0 + 8 + j) * KSPAN + key] = b[j];
        }
      } else {
#pragma unroll
        for (int j = 0; j < 16; ++j) Vt[(c0 + j) * KSPAN + key] = (half_t)0.f;
      }
    }
  }
  __syncthreads();

  // GEMM 1: scores[QB][KSPAN] = Q @ K^T.  2 x 10 = 20 16x16 tiles.
  for (int tt = wid; tt < (QB / 16) * (KSPAN / 16); tt += 8) {
    const int tm = tt % (QB / 16);
    const int tn = tt / (QB / 16);
    v8f acc = {};
#pragma unroll
    for (int kc = 0; kc < HD; kc += 32) {
      v16h a = load_a_frag(Qt, HD, tm * 16, kc);
      v16h b = load_bt_frag(Kt, HD, tn * 16, kc);
      acc = wmma_f16(a, b, acc);
    }
    const int lane = t & 31, l0 = lane & 15, hi = lane >> 4;
#pragma unroll
    for (int r = 0; r < 8; ++r)
      Sc[(tm * 16 + hi * 8 + r) * KSPAN + tn * 16 + l0] = (half_t)acc[r];
  }
  __syncthreads();

  // Softmax with sliding-window mask; probs written back in place (f16).
  if (t < QB) {
    const int qg = q0 + t;
    half_t* srow = Sc + t * KSPAN;
    float m = -1e30f;
    for (int c = 0; c < KSPAN; ++c) {
      const int kg = kg0 + c;
      const bool valid = (kg >= 0) & (kg < S) & (kg - qg <= 64) & (qg - kg <= 64);
      const float sv = valid ? (float)srow[c] : -1e30f;
      m = fmaxf(m, sv);
    }
    float sum = 0.f;
    for (int c = 0; c < KSPAN; ++c) {
      const int kg = kg0 + c;
      const bool valid = (kg >= 0) & (kg < S) & (kg - qg <= 64) & (qg - kg <= 64);
      const float e = valid ? __expf((float)srow[c] - m) : 0.f;
      sum += e;
      srow[c] = (half_t)e;
    }
    const float inv = 1.f / sum;
    for (int c = 0; c < KSPAN; ++c)
      srow[c] = (half_t)((float)srow[c] * inv);
  }
  __syncthreads();

  // GEMM 2: out[QB][HD] = P @ V.  2 x 4 = 8 tiles, one per wave.
  {
    const int tm = wid & 1;
    const int tn = wid >> 1;
    v8f acc = {};
#pragma unroll
    for (int kc = 0; kc < KSPAN; kc += 32) {
      v16h a = load_a_frag(Sc, KSPAN, tm * 16, kc);
      v16h b = load_bt_frag(Vt, KSPAN, tn * 16, kc);
      acc = wmma_f16(a, b, acc);
    }
    const int lane = t & 31, l0 = lane & 15, hi = lane >> 4;
#pragma unroll
    for (int r = 0; r < 8; ++r) {
      const int qg = q0 + tm * 16 + hi * 8 + r;
      attn[(size_t)qg * H + h * HD + tn * 16 + l0] = (half_t)acc[r];
    }
  }
}

// ---------------------------------------------------------------------------
// Kernel 3: out = attn @ Wo^T (4096x768), f32 output straight from C frags.
// Grid (S/64, H/64), 256 threads. LDS: A 64x32 f16 + B 64x32 f16 = 8 KB.
// ---------------------------------------------------------------------------
__global__ void __launch_bounds__(256) out_proj_kernel(
    const half_t* __restrict__ attn, const float* __restrict__ Wo,
    float* __restrict__ out) {
  extern __shared__ char smem[];
  half_t* At = (half_t*)smem;
  half_t* Bt = At + 64 * 32;

  const int m0 = blockIdx.x * 64;
  const int n0 = blockIdx.y * 64;
  const int t  = threadIdx.x;
  const int wid = t >> 5;
  const int wm = wid >> 1, wn = wid & 1;

  const int lrow = t >> 2;
  const int lcol = (t & 3) * 8;

  v8f acc0 = {}, acc1 = {};
  for (int kc = 0; kc < H; kc += 32) {
    __syncthreads();
    *(v8h*)(At + lrow * 32 + lcol) =
        *(const v8h*)(attn + (size_t)(m0 + lrow) * H + kc + lcol);
    {
      const float* sb = Wo + (size_t)(n0 + lrow) * H + kc + lcol;
      v4f b0 = *(const v4f*)sb;
      v4f b1 = *(const v4f*)(sb + 4);
      half_t* db = Bt + lrow * 32 + lcol;
#pragma unroll
      for (int j = 0; j < 4; ++j) { db[j] = (half_t)b0[j]; db[4 + j] = (half_t)b1[j]; }
      if (kc + 32 < H) __builtin_prefetch(sb + 32, 0, 0);
    }
    __syncthreads();
    v16h a  = load_a_frag(At, 32, wm * 16, 0);
    v16h b0 = load_bt_frag(Bt, 32, wn * 32, 0);
    v16h b1 = load_bt_frag(Bt, 32, wn * 32 + 16, 0);
    acc0 = wmma_f16(a, b0, acc0);
    acc1 = wmma_f16(a, b1, acc1);
  }
  const int lane = t & 31, l0 = lane & 15, hi = lane >> 4;
#pragma unroll
  for (int r = 0; r < 8; ++r) {
    const int m = m0 + wm * 16 + hi * 8 + r;
    out[(size_t)m * H + n0 + wn * 32 + l0]      = acc0[r];
    out[(size_t)m * H + n0 + wn * 32 + 16 + l0] = acc1[r];
  }
}

// ---------------------------------------------------------------------------
extern "C" void kernel_launch(void* const* d_in, const int* in_sizes, int n_in,
                              void* d_out, int out_size, void* d_ws, size_t ws_size,
                              hipStream_t stream) {
  (void)in_sizes; (void)n_in; (void)out_size; (void)ws_size;
  const float* x    = (const float*)d_in[0];
  const float* cosT = (const float*)d_in[1];
  const float* sinT = (const float*)d_in[2];
  const float* Wqkv = (const float*)d_in[3];
  const float* Wo   = (const float*)d_in[4];
  float* out = (float*)d_out;

  // Workspace: qkvh [3][NH][S][HD] f16 (18 MB) + attn [S][H] f16 (6 MB).
  half_t* qkvh = (half_t*)d_ws;
  half_t* attn = qkvh + (size_t)3 * NH * S * HD;

  const dim3 blk(256);

  // 1) QKV projection + RoPE
  {
    const size_t lds = (size_t)(64 * 32 + 64 * 32) * sizeof(half_t)
                     + (size_t)64 * 64 * sizeof(float);   // 24 KB
    qkv_rope_kernel<<<dim3(S / 64, (3 * H) / 64), blk, lds, stream>>>(
        x, Wqkv, cosT, sinT, qkvh);
  }
  // 2) Sliding-window attention
  {
    const size_t lds = (size_t)(QB * HD + KSPAN * HD + HD * KSPAN + QB * KSPAN)
                       * sizeof(half_t);                  // 54 KB
    attn_kernel<<<dim3(S / QB, NH), blk, lds, stream>>>(qkvh, attn);
  }
  // 3) Output projection
  {
    const size_t lds = (size_t)(2 * 64 * 32) * sizeof(half_t);  // 8 KB
    out_proj_kernel<<<dim3(S / 64, H / 64), blk, lds, stream>>>(attn, Wo, out);
  }
}
